// Reg2DDecode3D_87402584473879
// MI455X (gfx1250) — compile-verified, matrix operand order checked by reading, not run on hardware
//
#include <hip/hip_runtime.h>
#include <hip/hip_bf16.h>

typedef __attribute__((ext_vector_type(16))) _Float16 v16h;
typedef __attribute__((ext_vector_type(8)))  _Float16 v8h;
typedef __attribute__((ext_vector_type(2)))  _Float16 h2;
typedef __attribute__((ext_vector_type(8)))  float    v8f;

#define GEMM_TRANSB 1
#define GEMM_RELU   2
#define GEMM_RES    4

// =====================================================================
// Generic batched GEMM via CDNA5 WMMA, templated on B-transposition.
// D[b] = act( A[b](MxK) * B[b](KxN, or NxK row-major if TB) + bias + Res[b] )
// Block: 256 threads = 8 waves; tile 64(M) x 64(N); K stepped by 32.
// f32 -> f16 on the global->LDS path (packed h2 stores). LDS holds A as MxK
// f16 and B transposed as NxK f16 so each WMMA fragment is two contiguous
// 16B LDS reads (documented 16-bit 16x32 layout: lane<16 K=[0..7],[16..23];
// lane>=16 K=[8..15],[24..31]). Each wave owns 16(M) x 32(N):
// 1 shared A frag, 2 B frags, 2 WMMAs per K-step.
// Loader coordinates (rows, validity, LDS addresses) are hoisted out of the
// K loop; inner loop is branch-free clamp+select with 32-bit address math.
// =====================================================================
template<bool TB>
__global__ __launch_bounds__(256) void gemm_f16wmma_kernel(
    const float* __restrict__ A, const float* __restrict__ B,
    const float* __restrict__ bias, const float* __restrict__ Res,
    float* __restrict__ D, int M, int N, int K,
    long long sA, long long sB, long long sRes, long long sD,
    int lda, int ldb, int ldres, int ldd, int flags)
{
    A += (long long)blockIdx.z * sA;
    B += (long long)blockIdx.z * sB;
    D += (long long)blockIdx.z * sD;
    const float* R = Res ? (Res + (long long)blockIdx.z * sRes) : nullptr;

    __shared__ _Float16 sAh[64][32];   // M x K tile (f16)   4 KB
    __shared__ _Float16 sBh[64][32];   // N x K tile (f16)   4 KB

    const int tid    = threadIdx.x;
    const int lane   = tid & 31;
    const int wave   = tid >> 5;          // 0..7
    const int wm     = wave & 3;          // 0..3  -> 16-row M sub-tile
    const int wn     = wave >> 2;         // 0..1  -> 32-col N half
    const int m0     = blockIdx.x * 64;
    const int n0     = blockIdx.y * 64;
    const int laneHi = lane >> 4;
    const int lane15 = lane & 15;

    // ---- loader geometry, fixed across the whole K loop ----
    const int cp = tid & 15;              // k-pair index 0..15 (k = 2*cp, 2*cp+1)
    const int r0 = tid >> 4;              // 0..15; rows/cols r0 + 16*it
    const float* Arow[4]; bool okA[4];
    const float* Bbase[4]; bool okB[4];   // TB: row base; !TB: column base
    _Float16* sAp[4];
    _Float16* sBp[4];
#pragma unroll
    for (int it = 0; it < 4; ++it) {
        int rr = r0 + 16*it;
        int gm = m0 + rr;
        okA[it]  = gm < M;
        Arow[it] = A + (okA[it] ? gm : 0) * lda;
        int gn = n0 + rr;
        okB[it]  = gn < N;
        Bbase[it] = TB ? (B + (okB[it] ? gn : 0) * ldb)
                       : (B + (okB[it] ? gn : 0));
        sAp[it] = &sAh[rr][cp*2];
        sBp[it] = &sBh[rr][cp*2];
    }

    v8f acc0 = {0.f,0.f,0.f,0.f,0.f,0.f,0.f,0.f};
    v8f acc1 = {0.f,0.f,0.f,0.f,0.f,0.f,0.f,0.f};

    for (int k0 = 0; k0 < K; k0 += 32) {
        const int gk  = k0 + cp*2;
        const bool ok0 = gk     < K;
        const bool ok1 = gk + 1 < K;
        const int kc0 = ok0 ? gk     : 0;
        const int kc1 = ok1 ? gk + 1 : 0;
        const int bo0 = TB ? kc0 : kc0 * ldb;   // B offsets (32-bit)
        const int bo1 = TB ? kc1 : kc1 * ldb;

#pragma unroll
        for (int it = 0; it < 4; ++it) {
            float x0 = Arow[it][kc0];
            float x1 = Arow[it][kc1];
            x0 = (okA[it] && ok0) ? x0 : 0.f;
            x1 = (okA[it] && ok1) ? x1 : 0.f;
            h2 p; p[0] = (_Float16)x0; p[1] = (_Float16)x1;
            *(h2*)sAp[it] = p;
        }
#pragma unroll
        for (int it = 0; it < 4; ++it) {
            float x0 = Bbase[it][bo0];
            float x1 = Bbase[it][bo1];
            x0 = (okB[it] && ok0) ? x0 : 0.f;
            x1 = (okB[it] && ok1) ? x1 : 0.f;
            h2 p; p[0] = (_Float16)x0; p[1] = (_Float16)x1;
            *(h2*)sBp[it] = p;
        }
        __syncthreads();

        // ---- fragments: two contiguous 16B chunks each ----
        const _Float16* pa = &sAh[wm*16 + lane15][laneHi*8];
        v8h alo = *(const v8h*)pa;
        v8h ahi = *(const v8h*)(pa + 16);
        v16h af = __builtin_shufflevector(alo, ahi,
                    0,1,2,3,4,5,6,7,8,9,10,11,12,13,14,15);

        const _Float16* pb0 = &sBh[wn*32 + lane15][laneHi*8];
        v8h b0lo = *(const v8h*)pb0;
        v8h b0hi = *(const v8h*)(pb0 + 16);
        v16h bf0 = __builtin_shufflevector(b0lo, b0hi,
                    0,1,2,3,4,5,6,7,8,9,10,11,12,13,14,15);

        const _Float16* pb1 = &sBh[wn*32 + 16 + lane15][laneHi*8];
        v8h b1lo = *(const v8h*)pb1;
        v8h b1hi = *(const v8h*)(pb1 + 16);
        v16h bf1 = __builtin_shufflevector(b1lo, b1hi,
                    0,1,2,3,4,5,6,7,8,9,10,11,12,13,14,15);

        acc0 = __builtin_amdgcn_wmma_f32_16x16x32_f16(false, af, false, bf0,
                                                      (short)0, acc0, false, false);
        acc1 = __builtin_amdgcn_wmma_f32_16x16x32_f16(false, af, false, bf1,
                                                      (short)0, acc1, false, false);
        __syncthreads();
    }

    // ---- epilogue: C/D layout VGPR r -> M = r + 8*laneHi ; N = lane&15 ----
    const int nc0 = n0 + wn*32 + lane15;
    const int nc1 = nc0 + 16;
    const float bb0 = (bias && nc0 < N) ? bias[nc0] : 0.f;
    const float bb1 = (bias && nc1 < N) ? bias[nc1] : 0.f;
#pragma unroll
    for (int r = 0; r < 8; ++r) {
        int mm = m0 + wm*16 + r + 8*laneHi;
        if (mm < M) {
            if (nc0 < N) {
                float v = acc0[r] + bb0;
                if (flags & GEMM_RES)  v += R[(long long)mm*ldres + nc0];
                if (flags & GEMM_RELU) v = fmaxf(v, 0.f);
                D[(long long)mm*ldd + nc0] = v;
            }
            if (nc1 < N) {
                float v = acc1[r] + bb1;
                if (flags & GEMM_RES)  v += R[(long long)mm*ldres + nc1];
                if (flags & GEMM_RELU) v = fmaxf(v, 0.f);
                D[(long long)mm*ldd + nc1] = v;
            }
        }
    }
}

// =====================================================================
// Bilinear grid-sample (align_corners=True) fused with the uv transform.
// =====================================================================
__global__ void sample_kernel(const float* __restrict__ feat,
                              const float* __restrict__ uv,
                              float* __restrict__ out,
                              int C, int H, int W, int ldo)
{
    int i = blockIdx.x * blockDim.x + threadIdx.x;
    int total = 64 * 21 * C;
    if (i >= total) return;
    int c = i % C; int p = (i / C) % 21; int b = i / (C * 21);

    float u  = uv[(size_t)(b*21 + p)*2 + 0];
    float vv = uv[(size_t)(b*21 + p)*2 + 1];
    u  = fminf(fmaxf((u  - 0.5f) * 2.f, -1.f), 1.f);
    vv = fminf(fmaxf((vv - 0.5f) * 2.f, -1.f), 1.f);
    float gx = (u  + 1.f) * 0.5f * (float)(W - 1);
    float gy = (vv + 1.f) * 0.5f * (float)(H - 1);
    float fxf = floorf(gx), fyf = floorf(gy);
    int x0 = min(max((int)fxf, 0), W - 1);
    int y0 = min(max((int)fyf, 0), H - 1);
    int x1 = min(x0 + 1, W - 1);
    int y1 = min(y0 + 1, H - 1);
    float fx = gx - fxf, fy = gy - fyf;

    const float* fb = feat + ((size_t)b * C + c) * (size_t)(H * W);
    float v00 = fb[y0*W + x0], v01 = fb[y0*W + x1];
    float v10 = fb[y1*W + x0], v11 = fb[y1*W + x1];
    float r = v00*(1.f-fx)*(1.f-fy) + v01*fx*(1.f-fy)
            + v10*(1.f-fx)*fy       + v11*fx*fy;
    out[(size_t)(b*21 + p) * ldo + c] = r;
}

// ============ attention: one block per (t, head, batch), 64 threads ============
__global__ void attn_kernel(const float* __restrict__ q,
                            const float* __restrict__ k,
                            const float* __restrict__ v,
                            float* __restrict__ o)
{
    int t = blockIdx.x, h = blockIdx.y, b = blockIdx.z;
    int d = threadIdx.x;                   // 0..63
    __shared__ float sc[21];
    const float* qrow = q + ((size_t)(b*21 + t)*256 + h*64);
    if (d < 21) {
        const float* krow = k + ((size_t)(b*21 + d)*256 + h*64);
        float s = 0.f;
#pragma unroll
        for (int j = 0; j < 64; ++j) s += qrow[j] * krow[j];
        sc[d] = s * 0.125f;                // 1/sqrt(64)
    }
    __syncthreads();
    float mx = -1e30f;
    for (int s = 0; s < 21; ++s) mx = fmaxf(mx, sc[s]);
    float denom = 0.f, pv = 0.f;
    for (int s = 0; s < 21; ++s) {
        float e = __expf(sc[s] - mx);
        denom += e;
        pv += e * v[((size_t)(b*21 + s)*256 + h*64) + d];
    }
    o[((size_t)(b*21 + t)*256 + h*64) + d] = pv / denom;
}

// ============ LayerNorm over 256, one block per row ============
__global__ void ln_kernel(const float* __restrict__ x,
                          const float* __restrict__ g,
                          const float* __restrict__ b,
                          float* __restrict__ out)
{
    int row = blockIdx.x;
    int c = threadIdx.x;
    __shared__ float red[256];
    float val = x[(size_t)row*256 + c];
    red[c] = val; __syncthreads();
    for (int s = 128; s > 0; s >>= 1) { if (c < s) red[c] += red[c + s]; __syncthreads(); }
    float mean = red[0] * (1.f/256.f);
    __syncthreads();
    float dv = val - mean;
    red[c] = dv * dv; __syncthreads();
    for (int s = 128; s > 0; s >>= 1) { if (c < s) red[c] += red[c + s]; __syncthreads(); }
    float var = red[0] * (1.f/256.f);
    out[(size_t)row*256 + c] = dv * rsqrtf(var + 1e-5f) * g[c] + b[c];
}

// ============ mesh pool: row = repeat(arange,3) -> 3-tap gather ============
__global__ void pool_kernel(const float* __restrict__ z, const int* __restrict__ col,
                            const float* __restrict__ val, float* __restrict__ out,
                            int n_out, int n_in, int c)
{
    int i = blockIdx.x * blockDim.x + threadIdx.x;
    int total = 64 * n_out * c;
    if (i >= total) return;
    int ch = i % c; int r = (i / c) % n_out; int b = i / (c * n_out);
    float a = 0.f;
#pragma unroll
    for (int t = 0; t < 3; ++t) {
        int j = 3*r + t;
        a += val[j] * z[((size_t)b*n_in + col[j])*c + ch];
    }
    out[((size_t)b*n_out + r)*c + ch] = a;
}

// ============ spiral depthwise gather (9 taps) ============
__global__ void spiral_dw_kernel(const float* __restrict__ z, const int* __restrict__ idx,
                                 const float* __restrict__ dw, float* __restrict__ h,
                                 int n, int c)
{
    int i = blockIdx.x * blockDim.x + threadIdx.x;
    int total = 64 * n * c;
    if (i >= total) return;
    int ch = i % c; int nv = (i / c) % n; int b = i / (c * n);
    const int* row = idx + nv * 9;
    float acc = 0.f;
#pragma unroll
    for (int s = 0; s < 9; ++s)
        acc += z[((size_t)b*n + row[s])*c + ch] * dw[ch*9 + s];
    h[((size_t)b*n + nv)*c + ch] = acc;
}

// ============ strided column copy / elementwise add ============
__global__ void copy_cols_kernel(const float* __restrict__ src, float* __restrict__ dst,
                                 int rows, int cols, int lds, int ldd)
{
    int i = blockIdx.x * blockDim.x + threadIdx.x;
    if (i >= rows * cols) return;
    int c = i % cols, r = i / cols;
    dst[(size_t)r*ldd + c] = src[(size_t)r*lds + c];
}

__global__ void add_kernel(const float* __restrict__ a, const float* __restrict__ b,
                           float* __restrict__ o, int n)
{
    int i = blockIdx.x * blockDim.x + threadIdx.x;
    if (i < n) o[i] = a[i] + b[i];
}

// =====================================================================
// Host orchestration
// =====================================================================
static inline void gemm(hipStream_t st, const float* A, const float* B,
                        const float* bias, const float* Res, float* D,
                        int M, int N, int K,
                        long long sA, long long sB, long long sRes, long long sD,
                        int lda, int ldb, int ldres, int ldd, int flags, int batch)
{
    dim3 g((M + 63) / 64, (N + 63) / 64, batch);
    if (flags & GEMM_TRANSB)
        gemm_f16wmma_kernel<true><<<g, 256, 0, st>>>(A, B, bias, Res, D, M, N, K,
                                                     sA, sB, sRes, sD, lda, ldb, ldres, ldd, flags);
    else
        gemm_f16wmma_kernel<false><<<g, 256, 0, st>>>(A, B, bias, Res, D, M, N, K,
                                                      sA, sB, sRes, sD, lda, ldb, ldres, ldd, flags);
}

extern "C" void kernel_launch(void* const* d_in, const int* in_sizes, int n_in,
                              void* d_out, int out_size, void* d_ws, size_t ws_size,
                              hipStream_t stream)
{
    (void)in_sizes; (void)n_in; (void)out_size; (void)ws_size;
    auto F = [&](int i) { return (const float*)d_in[i]; };
    auto I = [&](int i) { return (const int*)d_in[i]; };
    // Input index map (setup_inputs order, dicts flattened in insertion order):
    // 0 uv | 1 x | 2 pre_out | 3 stack1 | 4 stack2 | 5 de_conv_w
    // 6+12L+{wq,wk,wv,wo,ln1g,ln1b,w1,b1,w2,b2,ln2g,ln2b} | 54 upsample
    // 55+3i {gcnA,gcnW,gcnb} | 67+2i {ds_dw,ds_pw} | 75 head_dw | 76 head_pw
    // 77 c1_w | 78 c1_b | 79..82 spirals[0..3] | 83+3l {row,col,val}

    float* ws = (float*)d_ws;
    size_t off = 0;
    auto wsbuf = [&](size_t n) { float* p = ws + off; off += n; return p; };
    float* XS   = wsbuf(344064);    // sampled latent (B,21,256)
    float* ATTX = wsbuf(344064);    // transformer running x
    float* Q    = wsbuf(344064);
    float* Kb   = wsbuf(344064);
    float* V    = wsbuf(344064);
    float* OATT = wsbuf(344064);
    float* T1   = wsbuf(344064);
    float* FFNH = wsbuf(1376256);   // (B,21,1024)
    float* CAT  = wsbuf(924672);    // (B,21,688)
    float* Y21  = wsbuf(344064);
    float* XR   = wsbuf(802816);    // (B,49,256)
    float* YR   = wsbuf(802816);
    float* Z0   = wsbuf(3194880);   // decode ping buffers (max (B,195,256))
    float* Z1   = wsbuf(3194880);
    float* Z2   = wsbuf(3194880);
    float* FINE = wsbuf(149376);    // (B,778,3)
    float* OUT  = (float*)d_out;

    // --- grid-sample all features (uv transform fused); cat columns:
    //     [0:48]=pre, [48:112]=s1, [112:176]=s2, [176:432]=lat, [432:688]=att
    { int t = 64*21*48;  sample_kernel<<<(t+255)/256,256,0,stream>>>(F(2), F(0), CAT+0,   48, 64, 64, 688); }
    { int t = 64*21*64;  sample_kernel<<<(t+255)/256,256,0,stream>>>(F(3), F(0), CAT+48,  64, 64, 64, 688); }
    { int t = 64*21*64;  sample_kernel<<<(t+255)/256,256,0,stream>>>(F(4), F(0), CAT+112, 64, 64, 64, 688); }
    { int t = 64*21*256; sample_kernel<<<(t+255)/256,256,0,stream>>>(F(1), F(0), XS,     256, 32, 32, 256); }

    // --- fused 1x1 conv: lat = XS @ de_conv_w^T  (M=1344,K=256,N=256)
    gemm(stream, XS, F(5), nullptr, nullptr, ATTX, 1344, 256, 256,
         0,0,0,0, 256, 256, 0, 256, GEMM_TRANSB, 1);
    copy_cols_kernel<<<(1344*256+255)/256,256,0,stream>>>(ATTX, CAT+176, 1344, 256, 256, 688);

    // --- 4-layer transformer on (1344, 256)
    for (int L = 0; L < 4; ++L) {
        int p = 6 + 12*L;
        gemm(stream, ATTX, F(p+0), nullptr, nullptr, Q,  1344, 256, 256, 0,0,0,0, 256,256,0,256, 0, 1);
        gemm(stream, ATTX, F(p+1), nullptr, nullptr, Kb, 1344, 256, 256, 0,0,0,0, 256,256,0,256, 0, 1);
        gemm(stream, ATTX, F(p+2), nullptr, nullptr, V,  1344, 256, 256, 0,0,0,0, 256,256,0,256, 0, 1);
        attn_kernel<<<dim3(21,4,64), 64, 0, stream>>>(Q, Kb, V, OATT);
        gemm(stream, OATT, F(p+3), nullptr, ATTX, T1, 1344, 256, 256,
             0,0,0,0, 256,256,256,256, GEMM_RES, 1);              // x + o@wo
        ln_kernel<<<1344,256,0,stream>>>(T1, F(p+4), F(p+5), ATTX);
        gemm(stream, ATTX, F(p+6), F(p+7), nullptr, FFNH, 1344, 1024, 256,
             0,0,0,0, 256,1024,0,1024, GEMM_RELU, 1);             // relu(x@w1+b1)
        gemm(stream, FFNH, F(p+8), F(p+9), ATTX, T1, 1344, 256, 1024,
             0,0,0,0, 1024,256,256,256, GEMM_RES, 1);             // x + h@w2+b2
        ln_kernel<<<1344,256,0,stream>>>(T1, F(p+10), F(p+11), ATTX);
    }
    copy_cols_kernel<<<(1344*256+255)/256,256,0,stream>>>(ATTX, CAT+432, 1344, 256, 256, 688);

    // --- y = cat @ c1_w^T + c1_b
    gemm(stream, CAT, F(77), F(78), nullptr, Y21, 1344, 256, 688,
         0,0,0,0, 688, 688, 0, 256, GEMM_TRANSB, 1);

    // --- mesh upsample (21 -> 49), batched over B with shared left matrix
    gemm(stream, F(54), ATTX, nullptr, nullptr, XR, 49, 256, 21,
         0, 21*256, 0, 49*256, 21, 256, 0, 256, 0, 64);
    gemm(stream, F(54), Y21, nullptr, nullptr, YR, 49, 256, 21,
         0, 21*256, 0, 49*256, 21, 256, 0, 256, 0, 64);

    // --- decoder: GCN -> pool -> spiral dsconv x4 + head
    auto decode_run = [&](const float* zin, float* pred) {
        static const int GV[4] = {49, 98, 195, 389};
        static const int GD[4] = {256, 256, 128, 64};
        static const int NO[4] = {98, 195, 389, 778};
        static const int CO[4] = {256, 128, 64, 32};
        float *B0 = Z0, *B1 = Z1, *B2 = Z2;
        const float* cur = zin;
        for (int i = 0; i < 4; ++i) {
            int v = GV[i], d = GD[i], no = NO[i], co = CO[i];
            int lvl = 3 - i;
            // g = A @ z   (batched, A shared)
            gemm(stream, F(55+3*i), cur, nullptr, nullptr, B0, v, d, v,
                 0, (long long)v*d, 0, (long long)v*d, v, d, 0, d, 0, 64);
            // h = relu(g @ W + b)
            gemm(stream, B0, F(56+3*i), F(57+3*i), nullptr, B1, 64*v, d, d,
                 0,0,0,0, d, d, 0, d, GEMM_RELU, 1);
            // pool (ups[3-i])
            { int t = 64*no*d;
              pool_kernel<<<(t+255)/256,256,0,stream>>>(B1, I(84+3*lvl), F(85+3*lvl), B2, no, v, d); }
            // spiral depthwise
            { int t = 64*no*d;
              spiral_dw_kernel<<<(t+255)/256,256,0,stream>>>(B2, I(79+lvl), F(67+2*i), B0, no, d); }
            // pointwise: relu(h @ pw^T)
            gemm(stream, B0, F(68+2*i), nullptr, nullptr, B1, 64*no, co, d,
                 0,0,0,0, d, d, 0, co, GEMM_TRANSB | GEMM_RELU, 1);
            cur = B1;
            float* t = B0; B0 = B2; B2 = B1; B1 = t;   // rotate, keep `cur` live
        }
        // head dsconv (no relu, no bias)
        { int t = 64*778*32;
          spiral_dw_kernel<<<(t+255)/256,256,0,stream>>>(cur, I(79), F(75), B0, 778, 32); }
        gemm(stream, B0, F(76), nullptr, nullptr, pred, 64*778, 3, 32,
             0,0,0,0, 32, 32, 0, 3, GEMM_TRANSB, 1);
    };

    decode_run(XR, OUT);        // pred_rough -> out[0 : 64*778*3]
    decode_run(YR, FINE);       // pred_fine
    add_kernel<<<(149376+255)/256,256,0,stream>>>(OUT, FINE, OUT + 149376, 149376);
}
